// GaussianKernel_44487271252802
// MI455X (gfx1250) — matrix operand order repaired and checked by reading, not compiled, not run on hardware
//
#include <hip/hip_runtime.h>

typedef __attribute__((ext_vector_type(16))) __bf16 v16bf;
typedef __attribute__((ext_vector_type(4)))  __bf16 bf16x4;
typedef __attribute__((ext_vector_type(8)))  float  v8f;

#define MROWS 8192
#define KDIM  512
#define LDA_S 40    // __bf16 per A-tile row in LDS (80B: 8B/16B-aligned chunks, bank spread)
#define LDB_S 136   // __bf16 per transposed-B-tile row in LDS (272B: 16B-aligned)

union Frag { int4 q[2]; v16bf v; };

// ---------------- kernel 1: zero atomic accumulators in ws ----------------
__global__ void gk_zero(float* __restrict__ ws) {
  int t = threadIdx.x;
  for (int i = t; i < 513; i += 256) ws[MROWS + i] = 0.f;   // colsum[512] + scale slot
}

// ---------------- kernel 2: per-row ||x||^2 and column sums ----------------
__global__ __launch_bounds__(256) void gk_rowstats(const float* __restrict__ X,
                                                   float* __restrict__ ws) {
  __shared__ float cs[KDIM];
  const int t = threadIdx.x;
  for (int i = t; i < KDIM; i += 256) cs[i] = 0.f;
  __syncthreads();
  const int lane = t & 31, w = t >> 5;
  const int base = blockIdx.x * 256;
  for (int j = 0; j < 32; ++j) {
    const int row = base + w + 8 * j;
    const float* xr = X + (size_t)row * KDIM;
    float acc = 0.f;
    #pragma unroll
    for (int k = 0; k < KDIM / 32; ++k) {
      const int c = lane + 32 * k;
      const float v = xr[c];
      acc += v * v;
      atomicAdd(&cs[c], v);                 // ds_add_f32, conflict-free across lanes
    }
    #pragma unroll
    for (int m = 16; m > 0; m >>= 1) acc += __shfl_xor(acc, m, 32);
    if (lane == 0) ws[row] = acc;
  }
  __syncthreads();
  for (int i = t; i < KDIM; i += 256) atomicAdd(&ws[MROWS + i], cs[i]);
}

// ---------------- kernel 3: sigma^2 closed form -> f = 1/(2 sigma^2) ----------------
__global__ void gk_finalize(float* __restrict__ ws) {
  __shared__ float red[256];
  const int t = threadIdx.x;
  float s = 0.f;
  for (int i = t; i < MROWS; i += 256) s += ws[i];
  red[t] = s; __syncthreads();
  for (int off = 128; off > 0; off >>= 1) { if (t < off) red[t] += red[t + off]; __syncthreads(); }
  const float sumSq = red[0];
  __syncthreads();
  const float c0 = ws[MROWS + t], c1 = ws[MROWS + 256 + t];
  red[t] = c0 * c0 + c1 * c1; __syncthreads();
  for (int off = 128; off > 0; off >>= 1) { if (t < off) red[t] += red[t + off]; __syncthreads(); }
  if (t == 0) {
    const float ssq = red[0];
    const float Mf = (float)MROWS;
    const float sumd2 = 2.f * Mf * sumSq - 2.f * ssq;     // sum(d2); clamp effect ~ 0
    const float mean = sumd2 / (Mf * Mf);
    const float sigma2 = 1.0f * mean;                      // ALPHA = 1.0
    ws[MROWS + KDIM] = 1.f / (2.f * sigma2);
  }
}

// ---------------- kernel 4: fused bf16-WMMA gram tile + gaussian epilogue ----------------
__global__ __launch_bounds__(256) void gk_gemm(const float* __restrict__ X,
                                               const float* __restrict__ ws,
                                               float* __restrict__ out) {
  __shared__ __bf16 lA[128 * LDA_S];   // A tile: 128 rows x 32 k (row-major bf16)
  __shared__ __bf16 lB[32 * LDB_S];    // B tile: 32 k x 128 n (transposed bf16)

  const int t     = threadIdx.x;
  const int lane  = t & 31;
  const int w     = t >> 5;
  const int waveM = w & 3;                     // 4 waves along M (32 rows each)
  const int waveN = w >> 2;                    // 2 waves along N (64 cols each)
  const int rowBase = blockIdx.y * 128;
  const int colBase = blockIdx.x * 128;
  const int laneN = lane & 15;
  const int hi    = lane >> 4;                 // 0/1 lane-half

  v8f acc[2][4];
  const v8f vz = {0.f, 0.f, 0.f, 0.f, 0.f, 0.f, 0.f, 0.f};
  #pragma unroll
  for (int a = 0; a < 2; ++a)
    #pragma unroll
    for (int b = 0; b < 4; ++b) acc[a][b] = vz;

  // software-pipelined global staging registers (next K-slab)
  float4 ra[4], rb[4];
  #pragma unroll
  for (int it = 0; it < 4; ++it) {
    const int idx = t + it * 256;
    const int rr  = idx >> 3;
    const int cg  = idx & 7;
    ra[it] = *(const float4*)(X + (size_t)(rowBase + rr) * KDIM + cg * 4);
    rb[it] = *(const float4*)(X + (size_t)(colBase + rr) * KDIM + cg * 4);
  }

  for (int kb = 0; kb < KDIM; kb += 32) {
    // ---- commit staged registers to LDS as bf16 (native cvt; B transposed) ----
    #pragma unroll
    for (int it = 0; it < 4; ++it) {
      const int idx = t + it * 256;
      const int rr  = idx >> 3;                // 0..127 (tile row)
      const int cg  = idx & 7;                 // float4 group within 32-wide k slab
      const bf16x4 pa = { (__bf16)ra[it].x, (__bf16)ra[it].y,
                          (__bf16)ra[it].z, (__bf16)ra[it].w };
      *(bf16x4*)&lA[rr * LDA_S + cg * 4] = pa;       // one ds_store_b64
      const int k0 = cg * 4;
      lB[(k0 + 0) * LDB_S + rr] = (__bf16)rb[it].x;
      lB[(k0 + 1) * LDB_S + rr] = (__bf16)rb[it].y;
      lB[(k0 + 2) * LDB_S + rr] = (__bf16)rb[it].z;
      lB[(k0 + 3) * LDB_S + rr] = (__bf16)rb[it].w;
    }
    __syncthreads();

    // ---- prefetch next K-slab into registers (overlaps with WMMA below) ----
    if (kb + 32 < KDIM) {
      #pragma unroll
      for (int it = 0; it < 4; ++it) {
        const int idx = t + it * 256;
        const int rr  = idx >> 3;
        const int cg  = idx & 7;
        ra[it] = *(const float4*)(X + (size_t)(rowBase + rr) * KDIM + (kb + 32) + cg * 4);
        rb[it] = *(const float4*)(X + (size_t)(colBase + rr) * KDIM + (kb + 32) + cg * 4);
      }
    }

    // ---- fragments ----
    Frag aF[2], bF[4];
    #pragma unroll
    for (int tm = 0; tm < 2; ++tm) {
      // A (16x32 bf16): lane holds row M=lane%16; lane<16 -> K{0..7,16..23}, lane>=16 -> K{8..15,24..31}
      const __bf16* pa = &lA[(waveM * 32 + tm * 16 + laneN) * LDA_S + hi * 8];
      aF[tm].q[0] = *(const int4*)(pa);
      aF[tm].q[1] = *(const int4*)(pa + 16);
    }
    #pragma unroll
    for (int tn = 0; tn < 4; ++tn) {
      // B (32x16 bf16): lane holds column N=lane%16; lane<16 -> K{0..7,16..23}, lane>=16 -> K{8..15,24..31}
      const __bf16* pb = &lB[waveN * 64 + tn * 16 + laneN];
      #pragma unroll
      for (int s = 0; s < 16; ++s)
        bF[tn].v[s] = pb[((s < 8) ? (hi * 8 + s) : (16 + hi * 8 + (s - 8))) * LDB_S];
    }

    // ---- 8 WMMAs per wave per K-step ----
    #pragma unroll
    for (int tm = 0; tm < 2; ++tm)
      #pragma unroll
      for (int tn = 0; tn < 4; ++tn)
        acc[tm][tn] = __builtin_amdgcn_wmma_f32_16x16x32_bf16(
            false, aF[tm].v, false, bF[tn].v, (short)0, acc[tm][tn], false, false);

    __syncthreads();   // LDS reads done before next iteration's stores
  }

  // ---- epilogue: d2 = sq_i + sq_j - 2 g ; out = exp(-d2 * f) ----
  const float fscale = ws[MROWS + KDIM];       // 1/(2 sigma^2)
  #pragma unroll
  for (int tm = 0; tm < 2; ++tm) {
    const float sqa = ws[rowBase + waveM * 32 + tm * 16 + laneN];   // row-sq, per-lane
    #pragma unroll
    for (int tn = 0; tn < 4; ++tn) {
      const int j = colBase + waveN * 64 + tn * 16 + laneN;
      const float sqb = ws[j];
      #pragma unroll
      for (int r = 0; r < 8; ++r) {
        const float rowSq = __shfl(sqa, r + hi * 8, 32);            // C/D: M = r + 8*hi
        const float g = acc[tm][tn][r];
        float d2 = rowSq + sqb - 2.f * g;
        d2 = fmaxf(d2, 0.f);
        const int i = rowBase + waveM * 32 + tm * 16 + r + hi * 8;
        out[(size_t)i * MROWS + j] = __expf(-d2 * fscale);
      }
    }
  }
}

extern "C" void kernel_launch(void* const* d_in, const int* in_sizes, int n_in,
                              void* d_out, int out_size, void* d_ws, size_t ws_size,
                              hipStream_t stream) {
  const float* X = (const float*)d_in[0];
  float* out = (float*)d_out;
  float* ws  = (float*)d_ws;    // layout: [0..8191] row sq | [8192..8703] colsum | [8704] scale

  gk_zero<<<1, 256, 0, stream>>>(ws);
  gk_rowstats<<<MROWS / 256, 256, 0, stream>>>(X, ws);
  gk_finalize<<<1, 256, 0, stream>>>(ws);
  dim3 grid(MROWS / 128, MROWS / 128);
  gk_gemm<<<grid, 256, 0, stream>>>(X, ws, out);
}
